// LocoTrack_79319456023415
// MI455X (gfx1250) — compile-verified, matrix operand order checked by reading, not run on hardware
//
#include <hip/hip_runtime.h>
#include <hip/hip_bf16.h>
#include <math.h>
#include <stdint.h>

typedef __attribute__((ext_vector_type(16))) __bf16 v16bf;
typedef __attribute__((ext_vector_type(8)))  __bf16 v8bf;
typedef __attribute__((ext_vector_type(8)))  float  v8f;
typedef int v4i_vs __attribute__((vector_size(16)));

#define SEQ   1024
#define BATCH 8
#define TDIM  512
#define TFFN  2048
#define NHEAD 8
#define HDIM  64
#define ROWS  (BATCH * SEQ)

// ---------------------------------------------------------------------------
// CDNA5 async global->LDS helpers (ASYNCcnt path), with portable fallbacks.
// Builtin signature (from toolchain diagnostic): first param is int4* in AS1,
// second is the LDS destination, then imm offset + imm cpol.
// ---------------------------------------------------------------------------
__device__ __forceinline__ void async_load_b128(const void* g, void* l) {
#if __has_builtin(__builtin_amdgcn_global_load_async_to_lds_b128)
  __builtin_amdgcn_global_load_async_to_lds_b128(
      (__attribute__((address_space(1))) v4i_vs*)(uintptr_t)g,
      (__attribute__((address_space(3))) v4i_vs*)(uintptr_t)l,
      0, 0);
#else
  unsigned lds = (unsigned)(uintptr_t)l;
  unsigned long long ga = (unsigned long long)(uintptr_t)g;
  asm volatile("global_load_async_to_lds_b128 %0, %1, off"
               :: "v"(lds), "v"(ga) : "memory");
#endif
}

__device__ __forceinline__ void wait_async0() {
#if __has_builtin(__builtin_amdgcn_s_wait_asynccnt)
  __builtin_amdgcn_s_wait_asynccnt(0);
#else
  asm volatile("s_wait_asynccnt 0x0" ::: "memory");
#endif
}

// Load a 16x32 bf16 "A-style" WMMA fragment (ISA 7.12.2 16-bit A layout):
// lane L -> row = row0 + (L&15); K elements: half0 -> K=ko..ko+7, half1 -> K=ko+16..ko+23,
// ko = 0 for lanes 0-15, 8 for lanes 16-31. Two 16-byte loads per lane.
__device__ __forceinline__ v16bf frag_ld16x32(const __bf16* base, int stride,
                                              int row0, int dd, int lane) {
  int r  = row0 + (lane & 15);
  int ko = (lane & 16) ? 8 : 0;
  const __bf16* p = base + (size_t)r * stride + dd + ko;
  v8bf lo = *reinterpret_cast<const v8bf*>(p);
  v8bf hi = *reinterpret_cast<const v8bf*>(p + 16);
  v16bf f;
#pragma unroll
  for (int i = 0; i < 8; ++i) { f[i] = lo[i]; f[i + 8] = hi[i]; }
  return f;
}

#define WMMA_BF16(a, b, c) \
  __builtin_amdgcn_wmma_f32_16x16x32_bf16(false, (a), false, (b), (short)0, (c), false, false)

// ---------------------------------------------------------------------------
// Tiled WMMA GEMM: C[M,N] = A[M,K] (bf16) @ W[K,N] (bf16) + bias, epilogue MODE
// MODE 0: fp32 store, 1: bf16 store, 2: fp32 += (residual add), 3: GELU -> bf16
// Block = 256 threads (8 waves, 4x2), block tile 128x64, wave tile 32x32
// (4 WMMA accumulators). A tile staged with double-buffered async-to-LDS;
// B tile staged synchronously transposed so fragments are contiguous-in-K.
// ---------------------------------------------------------------------------
template <int MODE>
__global__ __launch_bounds__(256) void gemm_wmma(
    const __bf16* __restrict__ A, const __bf16* __restrict__ W,
    const float* __restrict__ bias, void* __restrict__ out,
    int M, int N, int K) {
  __shared__ __align__(16) __bf16 As[2][128][40];  // [buf][m][k]
  __shared__ __align__(16) __bf16 Bs[2][64][40];   // [buf][n][k] (transposed)

  const int m0   = blockIdx.y * 128;
  const int n0   = blockIdx.x * 64;
  const int tid  = threadIdx.x;
  const int lane = tid & 31;
  const int wid  = tid >> 5;
  const int mw   = (wid >> 1) * 32;  // wave M offset in block tile (0..96)
  const int nw   = (wid & 1) * 32;   // wave N offset in block tile (0/32)

  const int arow = tid >> 1;          // 0..127
  const int acol = (tid & 1) * 16;    // 0/16
  const int bk   = tid >> 3;          // 0..31
  const int bn   = (tid & 7) * 8;     // 0..56

  auto issueA = [&](int buf, int k0) {
    const __bf16* gp = A + (size_t)(m0 + arow) * K + k0 + acol;
    async_load_b128(gp,     &As[buf][arow][acol]);
    async_load_b128(gp + 8, &As[buf][arow][acol + 8]);
  };
  auto stageB = [&](int buf, int k0) {
    v8bf vw = *reinterpret_cast<const v8bf*>(W + (size_t)(k0 + bk) * N + n0 + bn);
#pragma unroll
    for (int j = 0; j < 8; ++j) Bs[buf][bn + j][bk] = vw[j];
  };

  v8f acc00 = {}, acc01 = {}, acc10 = {}, acc11 = {};

  issueA(0, 0);
  stageB(0, 0);
  for (int k0 = 0; k0 < K; k0 += 32) {
    const int cur = (k0 >> 5) & 1;
    wait_async0();       // this wave's async tile stores visible in LDS
    __syncthreads();     // whole tile (all waves) ready; prev compute done
    if (k0 + 32 < K) {   // prefetch next tile into the other buffer
      issueA(cur ^ 1, k0 + 32);
      stageB(cur ^ 1, k0 + 32);
    }
    v16bf a0 = frag_ld16x32(&As[cur][0][0], 40, mw, 0, lane);
    v16bf a1 = frag_ld16x32(&As[cur][0][0], 40, mw + 16, 0, lane);
    v16bf b0 = frag_ld16x32(&Bs[cur][0][0], 40, nw, 0, lane);
    v16bf b1 = frag_ld16x32(&Bs[cur][0][0], 40, nw + 16, 0, lane);
    acc00 = WMMA_BF16(a0, b0, acc00);
    acc01 = WMMA_BF16(a0, b1, acc01);
    acc10 = WMMA_BF16(a1, b0, acc10);
    acc11 = WMMA_BF16(a1, b1, acc11);
  }

  // Epilogue: C/D layout — VGPR r: lanes 0-15 -> M=r, lanes 16-31 -> M=r+8; N=lane&15
  const int   nnA = n0 + nw + (lane & 15);
  const int   nnB = nnA + 16;
  const float bv0 = bias ? bias[nnA] : 0.f;
  const float bv1 = bias ? bias[nnB] : 0.f;
  const int   mh  = (lane & 16) ? 8 : 0;
#pragma unroll
  for (int r = 0; r < 8; ++r) {
    int    mA = m0 + mw + r + mh;
    int    mB = mA + 16;
    float  v00 = acc00[r] + bv0, v01 = acc01[r] + bv1;
    float  v10 = acc10[r] + bv0, v11 = acc11[r] + bv1;
    size_t i00 = (size_t)mA * N + nnA, i01 = (size_t)mA * N + nnB;
    size_t i10 = (size_t)mB * N + nnA, i11 = (size_t)mB * N + nnB;
    if (MODE == 0) {
      float* o = (float*)out;
      o[i00] = v00; o[i01] = v01; o[i10] = v10; o[i11] = v11;
    } else if (MODE == 1) {
      __bf16* o = (__bf16*)out;
      o[i00] = (__bf16)v00; o[i01] = (__bf16)v01;
      o[i10] = (__bf16)v10; o[i11] = (__bf16)v11;
    } else if (MODE == 2) {
      float* o = (float*)out;
      o[i00] += v00; o[i01] += v01; o[i10] += v10; o[i11] += v11;
    } else {
      __bf16* o = (__bf16*)out;
      o[i00] = (__bf16)(0.5f * v00 * (1.f + erff(v00 * 0.70710678f)));
      o[i01] = (__bf16)(0.5f * v01 * (1.f + erff(v01 * 0.70710678f)));
      o[i10] = (__bf16)(0.5f * v10 * (1.f + erff(v10 * 0.70710678f)));
      o[i11] = (__bf16)(0.5f * v11 * (1.f + erff(v11 * 0.70710678f)));
    }
  }
}

// ---------------------------------------------------------------------------
// LayerNorm over last dim (512): fp32 in -> bf16 out. One 256-thread block/row.
// ---------------------------------------------------------------------------
__global__ __launch_bounds__(256) void ln_kernel(
    const float* __restrict__ x, const float* __restrict__ s,
    const float* __restrict__ b, __bf16* __restrict__ out) {
  __shared__ float red[256];
  const int    row = blockIdx.x;
  const int    t   = threadIdx.x;
  const float* xr  = x + (size_t)row * TDIM;
  float e0 = xr[t], e1 = xr[t + 256];
  red[t] = e0 + e1;
  __syncthreads();
  for (int o = 128; o > 0; o >>= 1) { if (t < o) red[t] += red[t + o]; __syncthreads(); }
  float mean = red[0] * (1.f / TDIM);
  __syncthreads();
  float d0 = e0 - mean, d1 = e1 - mean;
  red[t] = d0 * d0 + d1 * d1;
  __syncthreads();
  for (int o = 128; o > 0; o >>= 1) { if (t < o) red[t] += red[t + o]; __syncthreads(); }
  float rstd = rsqrtf(red[0] * (1.f / TDIM) + 1e-5f);
  out[(size_t)row * TDIM + t]       = (__bf16)(d0 * rstd * s[t] + b[t]);
  out[(size_t)row * TDIM + t + 256] = (__bf16)(d1 * rstd * s[t + 256] + b[t + 256]);
}

// ---------------------------------------------------------------------------
// Attention: one wave per (batch, head, 16-query strip). WMMA for QK^T and PV.
// ALiBi: heads 0-3 causal with slope*(j-i), heads 4-7 anti-causal slope*(i-j).
// Dyn LDS: scores bf16 [16][1024] (32KB) + V^T tile [64][40] (5KB).
// ---------------------------------------------------------------------------
__global__ __launch_bounds__(32) void attn_kernel(
    const __bf16* __restrict__ Q, const __bf16* __restrict__ Kt,
    const __bf16* __restrict__ V, __bf16* __restrict__ O) {
  extern __shared__ __align__(16) char smem[];
  __bf16* sc = reinterpret_cast<__bf16*>(smem);                // 16 x SEQ
  __bf16* vt = reinterpret_cast<__bf16*>(smem + 16 * SEQ * 2); // 64 x 40

  const int    lane = threadIdx.x;
  const int    q0   = blockIdx.x * 16;
  const int    hh   = blockIdx.y;
  const int    bb   = blockIdx.z;
  const size_t base = (size_t)bb * SEQ * TDIM + hh * HDIM;

  const int   hs    = (hh < 4) ? hh : hh - 4;
  const float slope = __powf(24.f, -(float)(hs + 1) * 0.25f);
  const bool  fwd   = hh < 4;
  const int   mh    = (lane & 16) ? 8 : 0;
  const int   nn    = lane & 15;

  v16bf qf0 = frag_ld16x32(Q + base + (size_t)q0 * TDIM, TDIM, 0, 0, lane);
  v16bf qf1 = frag_ld16x32(Q + base + (size_t)q0 * TDIM, TDIM, 0, 32, lane);

  // scores strip: S_tile = Q(16x64) * K^T, + scale + ALiBi bias, -> LDS bf16
  for (int t0 = 0; t0 < SEQ; t0 += 16) {
    v16bf kf0 = frag_ld16x32(Kt + base + (size_t)t0 * TDIM, TDIM, 0, 0, lane);
    v16bf kf1 = frag_ld16x32(Kt + base + (size_t)t0 * TDIM, TDIM, 0, 32, lane);
    v8f s = {};
    s = WMMA_BF16(qf0, kf0, s);
    s = WMMA_BF16(qf1, kf1, s);
    const int j = t0 + nn;
#pragma unroll
    for (int r = 0; r < 8; ++r) {
      int   m = r + mh;
      int   i = q0 + m;
      float bias;
      if (fwd) bias = (j <= i) ? slope * (float)(j - i) : -1e9f;
      else     bias = (j >= i) ? slope * (float)(i - j) : -1e9f;
      sc[m * SEQ + j] = (__bf16)(s[r] * 0.125f + bias);
    }
  }
  __syncthreads();

  // softmax per row (32-lane cooperative)
  for (int r = 0; r < 16; ++r) {
    __bf16* rp = sc + r * SEQ;
    float   mx = -3.4e38f;
    for (int c = lane; c < SEQ; c += 32) mx = fmaxf(mx, (float)rp[c]);
#pragma unroll
    for (int o = 16; o > 0; o >>= 1) mx = fmaxf(mx, __shfl_xor(mx, o, 32));
    float sum = 0.f;
    for (int c = lane; c < SEQ; c += 32) {
      float e = __expf((float)rp[c] - mx);
      rp[c] = (__bf16)e;
      sum += e;
    }
#pragma unroll
    for (int o = 16; o > 0; o >>= 1) sum += __shfl_xor(sum, o, 32);
    float inv = 1.f / sum;
    for (int c = lane; c < SEQ; c += 32) rp[c] = (__bf16)((float)rp[c] * inv);
  }
  __syncthreads();

  // out(16x64) = P(16xSEQ) * V(SEQx64), V^T staged per 32-key tile
  v8f o0 = {}, o1 = {}, o2 = {}, o3 = {};
  for (int kt = 0; kt < SEQ; kt += 32) {
    const __bf16* vp = V + base + (size_t)(kt + lane) * TDIM;
#pragma unroll
    for (int c = 0; c < 8; ++c) {
      v8bf vv = *reinterpret_cast<const v8bf*>(vp + c * 8);
#pragma unroll
      for (int jd = 0; jd < 8; ++jd) vt[(c * 8 + jd) * 40 + lane] = vv[jd];
    }
    __syncthreads();
    v16bf pf  = frag_ld16x32(sc, SEQ, 0, kt, lane);
    v16bf vb0 = frag_ld16x32(vt, 40, 0, 0, lane);
    v16bf vb1 = frag_ld16x32(vt, 40, 16, 0, lane);
    v16bf vb2 = frag_ld16x32(vt, 40, 32, 0, lane);
    v16bf vb3 = frag_ld16x32(vt, 40, 48, 0, lane);
    o0 = WMMA_BF16(pf, vb0, o0);
    o1 = WMMA_BF16(pf, vb1, o1);
    o2 = WMMA_BF16(pf, vb2, o2);
    o3 = WMMA_BF16(pf, vb3, o3);
    __syncthreads();
  }
#pragma unroll
  for (int r = 0; r < 8; ++r) {
    int    srow = q0 + r + mh;
    size_t ob   = ((size_t)bb * SEQ + srow) * TDIM + hh * HDIM + nn;
    O[ob + 0]  = (__bf16)o0[r];
    O[ob + 16] = (__bf16)o1[r];
    O[ob + 32] = (__bf16)o2[r];
    O[ob + 48] = (__bf16)o3[r];
  }
}

__global__ void cvt_bf16(const float* __restrict__ in, __bf16* __restrict__ out, int n) {
  int i = blockIdx.x * blockDim.x + threadIdx.x;
  int st = gridDim.x * blockDim.x;
  for (; i < n; i += st) out[i] = (__bf16)in[i];
}

extern "C" void kernel_launch(void* const* d_in, const int* in_sizes, int n_in,
                              void* d_out, int out_size, void* d_ws, size_t ws_size,
                              hipStream_t stream) {
  const float* x     = (const float*)d_in[0];
  const float* w_in  = (const float*)d_in[1];
  const float* b_in  = (const float*)d_in[2];
  const float* wq    = (const float*)d_in[3];
  const float* bq    = (const float*)d_in[4];
  const float* wk    = (const float*)d_in[5];
  const float* bk    = (const float*)d_in[6];
  const float* wv    = (const float*)d_in[7];
  const float* bv    = (const float*)d_in[8];
  const float* wo    = (const float*)d_in[9];
  const float* bo    = (const float*)d_in[10];
  const float* ln1_s = (const float*)d_in[11];
  const float* ln1_b = (const float*)d_in[12];
  const float* ln2_s = (const float*)d_in[13];
  const float* ln2_b = (const float*)d_in[14];
  const float* w1    = (const float*)d_in[15];
  const float* b1    = (const float*)d_in[16];
  const float* w2    = (const float*)d_in[17];
  const float* b2    = (const float*)d_in[18];
  const float* lnf_s = (const float*)d_in[19];
  const float* lnf_b = (const float*)d_in[20];
  const float* w_out = (const float*)d_in[21];
  const float* b_out = (const float*)d_in[22];

  char*  ws  = (char*)d_ws;
  size_t off = 0;
  auto alloc_bf = [&](size_t e) { __bf16* p = (__bf16*)(ws + off); off += ((e * 2 + 255) & ~(size_t)255); return p; };
  auto alloc_f  = [&](size_t e) { float*  p = (float*)(ws + off);  off += ((e * 4 + 255) & ~(size_t)255); return p; };

  const size_t MR = ROWS;
  __bf16* xb     = alloc_bf(MR * TDIM);
  __bf16* hnb    = alloc_bf(MR * TDIM);
  __bf16* qb     = alloc_bf(MR * TDIM);
  __bf16* kbf    = alloc_bf(MR * TDIM);
  __bf16* vbf    = alloc_bf(MR * TDIM);
  __bf16* ab     = alloc_bf(MR * TDIM);
  __bf16* f1b    = alloc_bf(MR * TFFN);
  __bf16* w_in_b = alloc_bf((size_t)TDIM * TDIM);
  __bf16* wq_b   = alloc_bf((size_t)TDIM * TDIM);
  __bf16* wk_b   = alloc_bf((size_t)TDIM * TDIM);
  __bf16* wv_b   = alloc_bf((size_t)TDIM * TDIM);
  __bf16* wo_b   = alloc_bf((size_t)TDIM * TDIM);
  __bf16* w1_b   = alloc_bf((size_t)TDIM * TFFN);
  __bf16* w2_b   = alloc_bf((size_t)TFFN * TDIM);
  __bf16* wout_b = alloc_bf((size_t)TDIM * TDIM);
  float*  h      = alloc_f(MR * TDIM);

  auto cvt = [&](const float* src, __bf16* dst, int n) {
    int blocks = (n + 255) / 256;
    if (blocks > 4096) blocks = 4096;
    cvt_bf16<<<blocks, 256, 0, stream>>>(src, dst, n);
  };
  cvt(x, xb, (int)(MR * TDIM));
  cvt(w_in, w_in_b, TDIM * TDIM);
  cvt(wq, wq_b, TDIM * TDIM);
  cvt(wk, wk_b, TDIM * TDIM);
  cvt(wv, wv_b, TDIM * TDIM);
  cvt(wo, wo_b, TDIM * TDIM);
  cvt(w1, w1_b, TDIM * TFFN);
  cvt(w2, w2_b, TFFN * TDIM);
  cvt(w_out, wout_b, TDIM * TDIM);

  const dim3 g512(TDIM / 64, ROWS / 128);
  const dim3 gffn(TFFN / 64, ROWS / 128);

  // h = x @ w_in + b_in
  gemm_wmma<0><<<g512, 256, 0, stream>>>(xb, w_in_b, b_in, (void*)h, ROWS, TDIM, TDIM);
  // LN1
  ln_kernel<<<ROWS, 256, 0, stream>>>(h, ln1_s, ln1_b, hnb);
  // q/k/v projections (bf16 out)
  gemm_wmma<1><<<g512, 256, 0, stream>>>(hnb, wq_b, bq, (void*)qb, ROWS, TDIM, TDIM);
  gemm_wmma<1><<<g512, 256, 0, stream>>>(hnb, wk_b, bk, (void*)kbf, ROWS, TDIM, TDIM);
  gemm_wmma<1><<<g512, 256, 0, stream>>>(hnb, wv_b, bv, (void*)vbf, ROWS, TDIM, TDIM);
  // attention
  {
    dim3 ga(SEQ / 16, NHEAD, BATCH);
    size_t shmem = (size_t)16 * SEQ * 2 + (size_t)64 * 40 * 2;  // 37.8 KB
    attn_kernel<<<ga, 32, shmem, stream>>>(qb, kbf, vbf, ab);
  }
  // h += attn @ wo + bo
  gemm_wmma<2><<<g512, 256, 0, stream>>>(ab, wo_b, bo, (void*)h, ROWS, TDIM, TDIM);
  // LN2, FFN
  ln_kernel<<<ROWS, 256, 0, stream>>>(h, ln2_s, ln2_b, hnb);
  gemm_wmma<3><<<gffn, 256, 0, stream>>>(hnb, w1_b, b1, (void*)f1b, ROWS, TFFN, TDIM);
  gemm_wmma<2><<<g512, 256, 0, stream>>>(f1b, w2_b, b2, (void*)h, ROWS, TDIM, TFFN);
  // final LN + out projection (fp32 out)
  ln_kernel<<<ROWS, 256, 0, stream>>>(h, lnf_s, lnf_b, hnb);
  gemm_wmma<0><<<g512, 256, 0, stream>>>(hnb, wout_b, b_out, d_out, ROWS, TDIM, TDIM);
}
